// MixedEmbedderV2_10024453669503
// MI455X (gfx1250) — compile-verified, matrix operand order checked by reading, not run on hardware
//
#include <hip/hip_runtime.h>
#include <hip/hip_bf16.h>

typedef unsigned short u16;
typedef unsigned int u32;
typedef __attribute__((ext_vector_type(8)))  float  v8f;
typedef __attribute__((ext_vector_type(16))) __bf16 v16bf;
typedef __attribute__((ext_vector_type(4)))  u32    uint32x4;

union FragU { v16bf v; uint32x4 u[2]; };

// ---------------- problem constants ----------------
constexpr int BN   = 32768;
constexpr int FN   = 10;
constexpr int DN   = 512;
constexpr int HN   = 8;
constexpr int DHN  = 64;
constexpr int FFDN = 2048;
constexpr long long MROWS = (long long)BN * FN;   // 327680 (mult of 128)

// ---------------- small helpers ----------------
__device__ __forceinline__ u16 f2bf(float x) {
  union { float f; u32 u; } c; c.f = x;
  u32 u = c.u;
  u32 r = (u + 0x7FFFu + ((u >> 16) & 1u)) >> 16;
  return (u16)r;
}
__device__ __forceinline__ float bf2f(u16 h) {
  union { u32 u; float f; } c; c.u = ((u32)h) << 16;
  return c.f;
}
__device__ __forceinline__ float gelu_tanh(float x) {
  const float k0 = 0.7978845608028654f, k1 = 0.044715f;
  float t = tanhf(k0 * (x + k1 * x * x * x));
  return 0.5f * x * (1.0f + t);
}

__device__ __forceinline__ v8f wmma_bf16(v16bf a, v16bf b, v8f c) {
  return __builtin_amdgcn_wmma_f32_16x16x32_bf16(
      /*neg_a=*/false, a, /*neg_b=*/false, b,
      /*c_mod=*/(short)0, c, /*reuse_a=*/false, /*reuse_b=*/false);
}

// Load a 16x32 bf16 fragment (A-layout; also used for B from W^T rows).
// lane L: row = rowBase + (L&15); half = L>>4; ks = {k0+half*8 ..+7, k0+16+half*8 ..+7}
__device__ __forceinline__ v16bf load_frag(const u16* base, int ld, int rowBase,
                                           int k0, int lane) {
  const int m    = rowBase + (lane & 15);
  const int half = lane >> 4;
  const u16* p = base + (size_t)m * ld + (size_t)(k0 + half * 8);
  FragU f;
  f.u[0] = *(const uint32x4*)(p);
  f.u[1] = *(const uint32x4*)(p + 16);
  return f.v;
}

// ---------------- weight convert / transpose: fp32 [K][N] -> bf16 [N][K] ----
__global__ void convT_kernel(const float* __restrict__ in, u16* __restrict__ out,
                             int K, int N) {
  size_t i = (size_t)blockIdx.x * blockDim.x + threadIdx.x;
  if (i >= (size_t)K * N) return;
  int k = (int)(i / N);
  int n = (int)(i % N);
  out[(size_t)n * K + k] = f2bf(in[i]);
}

__global__ void pack_bias_kernel(const float* __restrict__ bq,
                                 const float* __restrict__ bk,
                                 const float* __restrict__ bv,
                                 float* __restrict__ out) {
  int i = blockIdx.x * blockDim.x + threadIdx.x;
  if (i >= 3 * DN) return;
  float v = (i < DN) ? bq[i] : (i < 2 * DN) ? bk[i - DN] : bv[i - 2 * DN];
  out[i] = v;
}

// ---------------- embedding + LN1 (wave per row) ----------------
__global__ void embed_ln_kernel(const float* __restrict__ cond,
                                const int* __restrict__ disc,
                                const float* __restrict__ emb,
                                const float* __restrict__ wc,
                                const float* __restrict__ bc,
                                const float* __restrict__ g,
                                const float* __restrict__ be,
                                float* __restrict__ x,
                                u16* __restrict__ h) {
  const int lane = threadIdx.x & 31;
  const int wid  = threadIdx.x >> 5;
  const size_t row = (size_t)blockIdx.x * 8 + wid;
  const int f = (int)(row % FN);
  const float c = cond[row];
  const int isd = disc[row];
  const int ci  = (int)c;

  float vals[16];
  float s = 0.f, s2 = 0.f;
#pragma unroll
  for (int i = 0; i < 16; ++i) {
    int d = lane + 32 * i;
    float v;
    if (isd > 0) v = emb[(size_t)ci * DN + d];
    else         v = c * wc[(size_t)f * DN + d] + bc[(size_t)f * DN + d];
    vals[i] = v; s += v; s2 += v * v;
  }
#pragma unroll
  for (int off = 16; off >= 1; off >>= 1) {
    s  += __shfl_xor(s,  off, 32);
    s2 += __shfl_xor(s2, off, 32);
  }
  float mu  = s * (1.0f / DN);
  float var = s2 * (1.0f / DN) - mu * mu;
  float r   = rsqrtf(var + 1e-5f);
#pragma unroll
  for (int i = 0; i < 16; ++i) {
    int d = lane + 32 * i;
    float nv = (vals[i] - mu) * r * g[d] + be[d];
    x[row * DN + d] = vals[i];
    h[row * DN + d] = f2bf(nv);
  }
}

// ---------------- generic LN: x fp32 -> bf16 or fp32 ----------------
template <bool BF16OUT>
__global__ void ln_kernel(const float* __restrict__ x,
                          const float* __restrict__ g,
                          const float* __restrict__ be,
                          u16* __restrict__ ob, float* __restrict__ of) {
  const int lane = threadIdx.x & 31;
  const int wid  = threadIdx.x >> 5;
  const size_t row = (size_t)blockIdx.x * 8 + wid;
  const float* xr = x + row * DN;
  float vals[16];
  float s = 0.f, s2 = 0.f;
#pragma unroll
  for (int i = 0; i < 16; ++i) {
    int d = lane + 32 * i;
    float v = xr[d];
    vals[i] = v; s += v; s2 += v * v;
  }
#pragma unroll
  for (int off = 16; off >= 1; off >>= 1) {
    s  += __shfl_xor(s,  off, 32);
    s2 += __shfl_xor(s2, off, 32);
  }
  float mu  = s * (1.0f / DN);
  float var = s2 * (1.0f / DN) - mu * mu;
  float r   = rsqrtf(var + 1e-5f);
#pragma unroll
  for (int i = 0; i < 16; ++i) {
    int d = lane + 32 * i;
    float nv = (vals[i] - mu) * r * g[d] + be[d];
    if (BF16OUT) ob[row * DN + d] = f2bf(nv);
    else         of[row * DN + d] = nv;
  }
}

// ---------------- WMMA GEMM: C = A[MxK] * Bt[NxK]^T + bias ----------------
// Workgroup: 256 threads (8 waves), tile 128(M) x 128(N); wave = 32x64 region
// (acc[2][4], 8 wmma per k-step from 2 A + 4 B fragment loads).
// MODE 0: store bf16 out[m*N+n];  MODE 1: fp32 residual add x[m*N+n] += val.
template <int MODE>
__global__ void __launch_bounds__(256, 1)
wmma_gemm_kernel(const u16* __restrict__ A,
                 const u16* __restrict__ Bt,
                 const float* __restrict__ bias,
                 u16* __restrict__ outBf,
                 float* __restrict__ outF32,
                 int N, int K) {
  const int lane = threadIdx.x & 31;
  const int wid  = threadIdx.x >> 5;
  const int mBase = blockIdx.y * 128 + (wid & 3) * 32;
  const int nBase = blockIdx.x * 128 + (wid >> 2) * 64;

  v8f zero = {};
  v8f acc[2][4];
#pragma unroll
  for (int mi = 0; mi < 2; ++mi)
#pragma unroll
    for (int nj = 0; nj < 4; ++nj) acc[mi][nj] = zero;

  for (int k0 = 0; k0 < K; k0 += 32) {
    if (k0 + 32 < K) {  // prefetch next k-step (global_prefetch_b8)
      __builtin_prefetch(A  + (size_t)(mBase + (lane & 15)) * K + k0 + 32, 0, 0);
      __builtin_prefetch(Bt + (size_t)(nBase + (lane & 15)) * K + k0 + 32, 0, 0);
    }
    v16bf a0 = load_frag(A, K, mBase,      k0, lane);
    v16bf a1 = load_frag(A, K, mBase + 16, k0, lane);
#pragma unroll
    for (int nj = 0; nj < 4; ++nj) {
      v16bf b = load_frag(Bt, K, nBase + nj * 16, k0, lane);
      acc[0][nj] = wmma_bf16(a0, b, acc[0][nj]);
      acc[1][nj] = wmma_bf16(a1, b, acc[1][nj]);
    }
  }

  const int half = lane >> 4;
  const int nIdx = lane & 15;
#pragma unroll
  for (int mi = 0; mi < 2; ++mi) {
#pragma unroll
    for (int nj = 0; nj < 4; ++nj) {
      int n = nBase + nj * 16 + nIdx;
      float bv = bias[n];
#pragma unroll
      for (int v = 0; v < 8; ++v) {
        int m = mBase + mi * 16 + v + 8 * half;
        float val = acc[mi][nj][v] + bv;
        size_t idx = (size_t)m * N + n;
        if (MODE == 0) outBf[idx] = f2bf(val);
        else           outF32[idx] += val;
      }
    }
  }
}

// ---------------- attention (F=10, tiny): thread per (b, h, qf) ------------
__global__ void __launch_bounds__(256, 1)
attn_kernel(const u16* __restrict__ qkv, u16* __restrict__ o) {
  long long t = (long long)blockIdx.x * blockDim.x + threadIdx.x;
  if (t >= (long long)BN * HN * FN) return;
  int qf = (int)(t % FN);
  int hh = (int)((t / FN) % HN);
  long long b = t / (FN * HN);

  const u16* qp = qkv + ((size_t)(b * FN + qf) * (3 * DN)) + hh * DHN;
  float q[DHN];
#pragma unroll
  for (int d = 0; d < DHN; ++d) q[d] = bf2f(qp[d]);

  float sc[FN];
  float mx = -1e30f;
#pragma unroll
  for (int kf = 0; kf < FN; ++kf) {
    const u16* kp = qkv + ((size_t)(b * FN + kf) * (3 * DN)) + DN + hh * DHN;
    float s = 0.f;
#pragma unroll
    for (int d = 0; d < DHN; ++d) s += q[d] * bf2f(kp[d]);
    s *= 0.125f;  // 1/sqrt(64)
    sc[kf] = s;
    mx = fmaxf(mx, s);
  }
  float se = 0.f;
#pragma unroll
  for (int kf = 0; kf < FN; ++kf) { sc[kf] = __expf(sc[kf] - mx); se += sc[kf]; }
  float inv = 1.0f / se;

  float out[DHN];
#pragma unroll
  for (int d = 0; d < DHN; ++d) out[d] = 0.f;
#pragma unroll
  for (int kf = 0; kf < FN; ++kf) {
    const u16* vp = qkv + ((size_t)(b * FN + kf) * (3 * DN)) + 2 * DN + hh * DHN;
    float w = sc[kf] * inv;
#pragma unroll
    for (int d = 0; d < DHN; ++d) out[d] += w * bf2f(vp[d]);
  }
  u16* op = o + (size_t)(b * FN + qf) * DN + hh * DHN;
#pragma unroll
  for (int d = 0; d < DHN; ++d) op[d] = f2bf(out[d]);
}

// ---------------- fused FFN: x += gelu(h@w1+b1)@w2 + b2 --------------------
// Workgroup: 256 threads, 32 rows.
// LDS: hTile[32][512] bf16 (32 KB) + u[32][2048] bf16 (128 KB) = 160 KB.
// Phase 1: wave w owns m-tile (w&1); its 16 A fragments are explicitly cached
//          in registers (128 VGPRs) and reused across 32 n-subtiles; processes
//          2 n-subtiles per iteration for load-latency overlap.
// Phase 2: K passed as runtime arg so the k-loop cannot be fully hoisted.
// __launch_bounds__(256,1): allow a full register allocation (no spills) —
// LDS already caps residency at 2 blocks/WGP.
__global__ void __launch_bounds__(256, 1)
ffn_fused_kernel(const u16* __restrict__ h,
                 const u16* __restrict__ w1T,
                 const float* __restrict__ b1,
                 const u16* __restrict__ w2T,
                 const float* __restrict__ b2,
                 float* __restrict__ x,
                 int Kffn) {
  extern __shared__ u16 uLds[];
  u16* hTile = uLds;                    // [32][DN]
  u16* uBuf  = uLds + 32 * DN;          // [32][FFDN]

  const int tid  = threadIdx.x;
  const int lane = tid & 31;
  const int wid  = tid >> 5;
  const int half = lane >> 4;
  const int nIdx = lane & 15;
  const long long rowBase = (long long)blockIdx.x * 32;

  // -------- stage h tile (32x512 bf16, coalesced b128) --------
  {
    const uint32x4* src = (const uint32x4*)(h + (size_t)rowBase * DN);
    uint32x4* dst = (uint32x4*)hTile;
#pragma unroll
    for (int j = 0; j < 8; ++j) dst[tid + 256 * j] = src[tid + 256 * j];
  }
  __syncthreads();

  const int mt = wid & 1;       // wave's fixed 16-row m-tile
  const int ng = wid >> 1;      // n-group 0..3

  // -------- phase 1: u = gelu(h @ w1 + b1) into LDS --------
  v16bf aF[16];
#pragma unroll
  for (int ks = 0; ks < 16; ++ks)
    aF[ks] = load_frag(hTile, DN, mt * 16, ks * 32, lane);

  for (int j = 0; j < 16; ++j) {        // 2 n-subtiles per iteration
    const int nt0 = ng * 32 + 2 * j;    // 0..127
    const int nt1 = nt0 + 1;
    v8f acc0 = {}, acc1 = {};
#pragma unroll
    for (int ks = 0; ks < 16; ++ks) {
      v16bf b0 = load_frag(w1T, DN, nt0 * 16, ks * 32, lane);
      v16bf b1f = load_frag(w1T, DN, nt1 * 16, ks * 32, lane);
      acc0 = wmma_bf16(aF[ks], b0, acc0);
      acc1 = wmma_bf16(aF[ks], b1f, acc1);
    }
#pragma unroll
    for (int p = 0; p < 2; ++p) {
      const int nt = p ? nt1 : nt0;
      const v8f& acc = p ? acc1 : acc0;
      const int n = nt * 16 + nIdx;
      const float bv = b1[n];
#pragma unroll
      for (int v = 0; v < 8; ++v) {
        int m = mt * 16 + v + 8 * half;
        uBuf[(size_t)m * FFDN + n] = f2bf(gelu_tanh(acc[v] + bv));
      }
    }
  }
  __syncthreads();

  // -------- phase 2: x += u @ w2 + b2  (K runtime => no mass hoist) --------
  for (int j = 0; j < 8; ++j) {
    const int nt = ng * 8 + j;           // 0..31
    v8f acc = {};
    for (int k0 = 0; k0 < Kffn; k0 += 32) {
      v16bf a = load_frag(uBuf, FFDN, mt * 16, k0, lane);   // ds_load_b128
      v16bf b = load_frag(w2T,  FFDN, nt * 16, k0, lane);
      acc = wmma_bf16(a, b, acc);
    }
    const int n = nt * 16 + nIdx;
    const float bv = b2[n];
#pragma unroll
    for (int v = 0; v < 8; ++v) {
      long long m = rowBase + mt * 16 + v + 8 * half;
      size_t idx = (size_t)m * DN + n;
      x[idx] = x[idx] + acc[v] + bv;
    }
  }
}

// ---------------- host orchestration ----------------
extern "C" void kernel_launch(void* const* d_in, const int* in_sizes, int n_in,
                              void* d_out, int out_size, void* d_ws, size_t ws_size,
                              hipStream_t stream) {
  const float* cond   = (const float*)d_in[0];
  const int*   disc   = (const int*)  d_in[1];
  const float* emb    = (const float*)d_in[2];
  const float* w_cont = (const float*)d_in[3];
  const float* b_cont = (const float*)d_in[4];
  const float* ln1_g  = (const float*)d_in[5];
  const float* ln1_b  = (const float*)d_in[6];
  const float* wq     = (const float*)d_in[7];
  const float* bq     = (const float*)d_in[8];
  const float* wk     = (const float*)d_in[9];
  const float* bk     = (const float*)d_in[10];
  const float* wv     = (const float*)d_in[11];
  const float* bv     = (const float*)d_in[12];
  const float* wo     = (const float*)d_in[13];
  const float* bo     = (const float*)d_in[14];
  const float* ln2_g  = (const float*)d_in[15];
  const float* ln2_b  = (const float*)d_in[16];
  const float* w1     = (const float*)d_in[17];
  const float* b1     = (const float*)d_in[18];
  const float* w2     = (const float*)d_in[19];
  const float* b2     = (const float*)d_in[20];
  const float* lnf_g  = (const float*)d_in[21];
  const float* lnf_b  = (const float*)d_in[22];

  char* ws = (char*)d_ws;
  constexpr size_t SZ_WQKVT = (size_t)3 * DN * DN * 2;
  constexpr size_t SZ_WOT   = (size_t)DN * DN * 2;
  constexpr size_t SZ_W1T   = (size_t)FFDN * DN * 2;
  constexpr size_t SZ_W2T   = (size_t)DN * FFDN * 2;
  constexpr size_t SZ_BQKV  = (size_t)3 * DN * 4;
  constexpr size_t SZ_X     = (size_t)MROWS * DN * 4;
  constexpr size_t SZ_H     = (size_t)MROWS * DN * 2;
  constexpr size_t SZ_QKV   = (size_t)MROWS * 3 * DN * 2;

  size_t off = 0;
  u16*   wqkvT = (u16*)(ws + off);   off += SZ_WQKVT;
  u16*   woT   = (u16*)(ws + off);   off += SZ_WOT;
  u16*   w1T   = (u16*)(ws + off);   off += SZ_W1T;
  u16*   w2T   = (u16*)(ws + off);   off += SZ_W2T;
  float* bqkv  = (float*)(ws + off); off += SZ_BQKV;
  off = (off + 255) & ~(size_t)255;
  float* x     = (float*)(ws + off); off += SZ_X;
  u16*   h     = (u16*)(ws + off);   off += SZ_H;
  u16*   qkv   = (u16*)(ws + off);   off += SZ_QKV;
  u16*   o     = (u16*)(ws + off);   off += SZ_H;
  (void)ws_size; (void)n_in; (void)in_sizes; (void)out_size;

  const dim3 blk(256);

  // 1) weight convert + transpose (fp32 [K][N] -> bf16 [N][K])
  convT_kernel<<<(DN * DN + 255) / 256, blk, 0, stream>>>(wq, wqkvT, DN, DN);
  convT_kernel<<<(DN * DN + 255) / 256, blk, 0, stream>>>(wk, wqkvT + (size_t)DN * DN, DN, DN);
  convT_kernel<<<(DN * DN + 255) / 256, blk, 0, stream>>>(wv, wqkvT + (size_t)2 * DN * DN, DN, DN);
  convT_kernel<<<(DN * DN + 255) / 256, blk, 0, stream>>>(wo, woT, DN, DN);
  convT_kernel<<<(DN * FFDN + 255) / 256, blk, 0, stream>>>(w1, w1T, DN, FFDN);
  convT_kernel<<<(DN * FFDN + 255) / 256, blk, 0, stream>>>(w2, w2T, FFDN, DN);
  pack_bias_kernel<<<(3 * DN + 255) / 256, blk, 0, stream>>>(bq, bk, bv, bqkv);

  // 2) mixed embedding + LN1 -> x (fp32), h (bf16)
  embed_ln_kernel<<<(unsigned)(MROWS / 8), blk, 0, stream>>>(
      cond, disc, emb, w_cont, b_cont, ln1_g, ln1_b, x, h);

  // 3) fused QKV GEMM: qkv = h @ [wq|wk|wv] + [bq|bk|bv]   (N=1536, K=512)
  {
    dim3 grid((3 * DN) / 128, (unsigned)(MROWS / 128));
    wmma_gemm_kernel<0><<<grid, blk, 0, stream>>>(h, wqkvT, bqkv, qkv, nullptr,
                                                  3 * DN, DN);
  }

  // 4) attention -> o (bf16)
  {
    long long nthr = (long long)BN * HN * FN;
    attn_kernel<<<(unsigned)((nthr + 255) / 256), blk, 0, stream>>>(qkv, o);
  }

  // 5) x += o @ wo + bo
  {
    dim3 grid(DN / 128, (unsigned)(MROWS / 128));
    wmma_gemm_kernel<1><<<grid, blk, 0, stream>>>(o, woT, bo, nullptr, x, DN, DN);
  }

  // 6) LN2 -> h (bf16)
  ln_kernel<true><<<(unsigned)(MROWS / 8), blk, 0, stream>>>(x, ln2_g, ln2_b, h, nullptr);

  // 7) fused FFN: x += gelu(h@w1+b1)@w2 + b2   (160 KB dynamic LDS)
  {
    const size_t lds = (size_t)(32 * DN + 32 * FFDN) * sizeof(u16);  // 163840
    hipFuncSetAttribute((const void*)ffn_fused_kernel,
                        hipFuncAttributeMaxDynamicSharedMemorySize, (int)lds);
    ffn_fused_kernel<<<(unsigned)(MROWS / 32), blk, lds, stream>>>(
        h, w1T, b1, w2T, b2, x, FFDN);
  }

  // 8) final LN -> d_out (fp32)
  ln_kernel<false><<<(unsigned)(MROWS / 8), blk, 0, stream>>>(
      x, lnf_g, lnf_b, nullptr, (float*)d_out);
}
